// GraphTrans_43164421324863
// MI455X (gfx1250) — matrix operand order repaired and checked by reading, not compiled, not run on hardware
//
#include <hip/hip_runtime.h>
#include <math.h>

// ---------------------------------------------------------------------------
// GraphTrans forward for MI455X (gfx1250).
// Dense GEMMs: one-time fp32->bf16 pack of A (row-major) and B (transposed),
// then a 64x64-macro-tile WMMA kernel whose hot loop is pure data movement
// (async global->LDS b128 copies, ASYNCcnt-tracked) + v_wmma_f32_16x16x32_bf16.
// Reductions, softmax, norms, scatter-aggregation stay fp32 VALU.
// Workspace requirement: ~152 MB (see layout below).
// ---------------------------------------------------------------------------

typedef __bf16 bf16_t;
typedef __attribute__((ext_vector_type(16))) bf16_t v16bf;
typedef __attribute__((ext_vector_type(8)))  bf16_t v8bf;
typedef __attribute__((ext_vector_type(4)))  bf16_t v4bf;
typedef __attribute__((ext_vector_type(8)))  float  v8f;
typedef __attribute__((ext_vector_type(4)))  float  f4;
typedef int gv4i __attribute__((vector_size(16)));   // matches builtin param type

union BFrag { v16bf v; bf16_t e[16]; };
union FAcc  { v8f   v; float  e[8];  };
union Pack4 { v4bf  v; bf16_t e[4];  };

// Async global->LDS 16-byte copy (ISA 15.18 GLOBAL_LOAD_ASYNC_TO_LDS_B128,
// tracked by ASYNCcnt). Guarded: falls back to b128 load + ds_store_b128.
#if __has_builtin(__builtin_amdgcn_global_load_async_to_lds_b128)
  #define ASYNC_COPY16(srcp, dstp)                                            \
      __builtin_amdgcn_global_load_async_to_lds_b128(                         \
          (__attribute__((address_space(1))) gv4i*)(srcp),                    \
          (__attribute__((address_space(3))) gv4i*)(dstp), 0, 0)
  #if __has_builtin(__builtin_amdgcn_s_wait_asynccnt)
    #define ASYNC_WAIT() __builtin_amdgcn_s_wait_asynccnt(0)
  #else
    #define ASYNC_WAIT() asm volatile("s_wait_asynccnt 0x0" ::: "memory")
  #endif
#else
  #define ASYNC_COPY16(srcp, dstp) (*(v8bf*)(dstp) = *(const v8bf*)(srcp))
  #define ASYNC_WAIT() ((void)0)
#endif

#define NTOK  7936          // 128*62 tokens / stage-0 nodes
#define MG    128           // graphs
#define NN_   62
#define DD    512
#define HH    8
#define DFF_  2048
#define LL    4
#define EG    76800         // 128*600 edges
#define KP1   50
#define KP2   40
#define N1_   (MG*KP1)      // 6400
#define N2_   (MG*KP2)      // 5120

// ---------------------------------------------------------------------------
// fp32 -> bf16 pack (vectorized), n4 = element count / 4
// ---------------------------------------------------------------------------
__global__ void cvt_bf16_kernel(const float* __restrict__ in, bf16_t* __restrict__ out, int n4)
{
    int i = blockIdx.x * 256 + threadIdx.x;
    if (i >= n4) return;
    f4 v = *(const f4*)(in + (size_t)i * 4);
    Pack4 p;
    p.e[0] = (bf16_t)v[0]; p.e[1] = (bf16_t)v[1];
    p.e[2] = (bf16_t)v[2]; p.e[3] = (bf16_t)v[3];
    *(v4bf*)(out + (size_t)i * 4) = p.v;
}

// fp32 B[K,N] -> bf16 Bt[N,K] (coalesced writes; strided reads are L2-hot)
__global__ void cvt_bf16_T_kernel(const float* __restrict__ B, bf16_t* __restrict__ Bt,
                                  int K, int N)
{
    int idx = blockIdx.x * 256 + threadIdx.x;
    if (idx >= N * K) return;
    int n = idx / K, k = idx % K;
    Bt[idx] = (bf16_t)B[(size_t)k * N + n];
}

// ---------------------------------------------------------------------------
// Fast path: C[M,N] = A[M,K] @ B[K,N] (+bias)(+relu), M%64==N%64==K%32==0.
// A is bf16 [M,K]; Bt is bf16 [N,K]. Block = 128 threads (4 waves), macro
// tile 64x64; per k-step each wave does a 32x32 quadrant = 4 WMMAs.
// Staging = pure 16B async copies; fragments = contiguous ds_load_b128.
// ---------------------------------------------------------------------------
__global__ __launch_bounds__(128)
void wmma_gemm64_kernel(const bf16_t* __restrict__ A, const bf16_t* __restrict__ Bt,
                        const float* __restrict__ bias, float* __restrict__ C,
                        int M, int N, int K, int relu)
{
    __shared__ __align__(16) bf16_t lA[64][40];   // [m][k], 80B rows
    __shared__ __align__(16) bf16_t lBt[64][40];  // [n][k]
    const int tid  = threadIdx.x;
    const int lane = tid & 31;
    const int wave = tid >> 5;
    const int wm = (wave >> 1) * 32;
    const int wn = (wave & 1) * 32;
    const int tn0 = blockIdx.x * 64;
    const int tm0 = blockIdx.y * 64;

    FAcc acc[2][2];
#pragma unroll
    for (int i = 0; i < 2; ++i)
#pragma unroll
        for (int j = 0; j < 2; ++j)
#pragma unroll
            for (int r = 0; r < 8; ++r) acc[i][j].e[r] = 0.0f;

    const int l16 = lane & 15;
    const int kA  = (lane < 16) ? 0 : 8;    // A fragment k-base (ISA 7.12.2)
    const int kB  = (lane < 16) ? 0 : 8;    // B fragment k-base

    for (int k0 = 0; k0 < K; k0 += 32) {
        // stage A (64x32) and Bt (64x32): 256 16B chunks each, 2 per thread
#pragma unroll
        for (int p = 0; p < 2; ++p) {
            const int idx8 = tid + p * 128;
            const int r  = idx8 >> 2;         // 0..63
            const int c8 = (idx8 & 3) << 3;   // 0,8,16,24
            ASYNC_COPY16(A  + (size_t)(tm0 + r) * K + (k0 + c8), &lA[r][c8]);
            ASYNC_COPY16(Bt + (size_t)(tn0 + r) * K + (k0 + c8), &lBt[r][c8]);
        }
        ASYNC_WAIT();
        __syncthreads();

        BFrag af[2], bfr[2];
#pragma unroll
        for (int i = 0; i < 2; ++i) {
            const int m = wm + i * 16 + l16;
            *(v8bf*)&af[i].e[0] = *(const v8bf*)&lA[m][kA];
            *(v8bf*)&af[i].e[8] = *(const v8bf*)&lA[m][kA + 16];
            const int n = wn + i * 16 + l16;
            *(v8bf*)&bfr[i].e[0] = *(const v8bf*)&lBt[n][kB];
            *(v8bf*)&bfr[i].e[8] = *(const v8bf*)&lBt[n][kB + 16];
        }
#pragma unroll
        for (int i = 0; i < 2; ++i)
#pragma unroll
            for (int j = 0; j < 2; ++j)
                acc[i][j].v = __builtin_amdgcn_wmma_f32_16x16x32_bf16(
                    false, af[i].v, false, bfr[j].v, (short)0, acc[i][j].v, false, false);
        __syncthreads();
    }

#pragma unroll
    for (int j = 0; j < 2; ++j) {
        const int col = tn0 + wn + j * 16 + l16;
        const float bv = bias ? bias[col] : 0.0f;
#pragma unroll
        for (int i = 0; i < 2; ++i) {
            const int rbase = tm0 + wm + i * 16 + ((lane < 16) ? 0 : 8);
#pragma unroll
            for (int r = 0; r < 8; ++r) {
                float v = acc[i][j].e[r] + bv;
                if (relu) v = fmaxf(v, 0.0f);
                C[(size_t)(rbase + r) * N + col] = v;
            }
        }
    }
}

// ---------------------------------------------------------------------------
// Edge path (ragged K=62 / N=62 GEMMs only): fp32 inputs, guarded 32x32.
// ---------------------------------------------------------------------------
__global__ __launch_bounds__(128)
void wmma_gemm_edge_kernel(const float* __restrict__ A, const float* __restrict__ B,
                           const float* __restrict__ bias, float* __restrict__ C,
                           int M, int N, int K, int relu)
{
    __shared__ __align__(16) bf16_t lA[32][40];
    __shared__ __align__(16) bf16_t lBt[32][40];
    const int tid  = threadIdx.x;
    const int lane = tid & 31;
    const int wave = tid >> 5;
    const int wm = (wave >> 1) * 16, wn = (wave & 1) * 16;
    const int tn0 = blockIdx.x * 32;
    const int tm0 = blockIdx.y * 32;

    FAcc acc;
#pragma unroll
    for (int i = 0; i < 8; ++i) acc.e[i] = 0.0f;

    const int l16 = lane & 15;
    const int kA  = (lane < 16) ? 0 : 8;
    const int kB  = (lane < 16) ? 0 : 8;

    for (int k0 = 0; k0 < K; k0 += 32) {
        float va[8], vb[8];
#pragma unroll
        for (int p = 0; p < 8; ++p) {
            const int idx = tid + p * 128;
            const int r = idx >> 5, c = idx & 31;
            const int gm = tm0 + r, gk = k0 + c;
            va[p] = (gm < M && gk < K) ? A[(size_t)gm * K + gk] : 0.0f;
            const int gk2 = k0 + r, gn = tn0 + c;
            vb[p] = (gk2 < K && gn < N) ? B[(size_t)gk2 * N + gn] : 0.0f;
        }
#pragma unroll
        for (int p = 0; p < 8; ++p) {
            const int idx = tid + p * 128;
            const int r = idx >> 5, c = idx & 31;
            lA[r][c]  = (bf16_t)va[p];
            lBt[c][r] = (bf16_t)vb[p];
        }
        __syncthreads();

        BFrag a, b;
        *(v8bf*)&a.e[0] = *(const v8bf*)&lA[wm + l16][kA];
        *(v8bf*)&a.e[8] = *(const v8bf*)&lA[wm + l16][kA + 16];
        *(v8bf*)&b.e[0] = *(const v8bf*)&lBt[wn + l16][kB];
        *(v8bf*)&b.e[8] = *(const v8bf*)&lBt[wn + l16][kB + 16];
        acc.v = __builtin_amdgcn_wmma_f32_16x16x32_bf16(
            false, a.v, false, b.v, (short)0, acc.v, false, false);
        __syncthreads();
    }

    const int col = tn0 + wn + l16;
    if (col < N) {
        const float bv = bias ? bias[col] : 0.0f;
        const int rbase = tm0 + wm + ((lane < 16) ? 0 : 8);
#pragma unroll
        for (int r = 0; r < 8; ++r) {
            const int row = rbase + r;
            if (row < M) {
                float v = acc.e[r] + bv;
                if (relu) v = fmaxf(v, 0.0f);
                C[(size_t)row * N + col] = v;
            }
        }
    }
}

// ------------------------- BatchNorm1d(62), train-mode stats ---------------
__global__ __launch_bounds__(256)
void bn_stats_kernel(const float* __restrict__ x, float* __restrict__ mu,
                     float* __restrict__ rsd)
{
    const int c = blockIdx.x, tid = threadIdx.x;
    __shared__ float rs[256], rq[256];
    float s = 0.f, q = 0.f;
    for (int idx = tid; idx < MG * NN_; idx += 256) {
        int bb = idx / NN_, j = idx % NN_;
        float v = x[((size_t)bb * NN_ + c) * NN_ + j];
        s += v; q += v * v;
    }
    rs[tid] = s; rq[tid] = q; __syncthreads();
    for (int off = 128; off > 0; off >>= 1) {
        if (tid < off) { rs[tid] += rs[tid + off]; rq[tid] += rq[tid + off]; }
        __syncthreads();
    }
    if (tid == 0) {
        float m = rs[0] / (float)(MG * NN_);
        float var = rq[0] / (float)(MG * NN_) - m * m;
        mu[c] = m; rsd[c] = rsqrtf(var + 1e-5f);
    }
}

// normalize + transpose(1,0,2): Axb row r=s*128+b  <-  x[(b*62+s)*62 + j]
__global__ void bn_apply_kernel(const float* __restrict__ x, const float* __restrict__ mu,
                                const float* __restrict__ rsd, const float* __restrict__ g,
                                const float* __restrict__ b, float* __restrict__ Axb)
{
    int idx = blockIdx.x * 256 + threadIdx.x;
    if (idx >= NTOK * NN_) return;
    int r = idx / NN_, j = idx % NN_;
    int s = r >> 7, bb = r & 127;
    float v = x[((size_t)bb * NN_ + s) * NN_ + j];
    Axb[idx] = (v - mu[s]) * rsd[s] * g[s] + b[s];
}

// ------------------------- attention (per s,b,h) ---------------------------
__global__ __launch_bounds__(64)
void attn_kernel(const float* __restrict__ qkv, float* __restrict__ o)
{
    const int s = blockIdx.x, b = blockIdx.y, h = blockIdx.z;
    const int tid = threadIdx.x;
    __shared__ float qs[64], p[64], red[64];
    const size_t rowQ = (size_t)(s * MG + b) * (3 * DD) + h * 64;
    qs[tid] = qkv[rowQ + tid];
    __syncthreads();
    float sc = -1e30f;
    if (tid < NN_) {
        const float* kr = qkv + (size_t)(tid * MG + b) * (3 * DD) + DD + h * 64;
        float d = 0.f;
#pragma unroll 8
        for (int i = 0; i < 64; ++i) d += qs[i] * kr[i];
        sc = d * 0.125f;
    }
    red[tid] = sc; __syncthreads();
    for (int off = 32; off > 0; off >>= 1) {
        if (tid < off) red[tid] = fmaxf(red[tid], red[tid + off]);
        __syncthreads();
    }
    const float mx = red[0]; __syncthreads();
    const float e = __expf(sc - mx);   // tid>=62 -> 0
    p[tid] = e; red[tid] = e; __syncthreads();
    for (int off = 32; off > 0; off >>= 1) {
        if (tid < off) red[tid] += red[tid + off];
        __syncthreads();
    }
    const float inv = 1.0f / red[0];
    float accv = 0.f;
    for (int t = 0; t < NN_; ++t)
        accv += p[t] * qkv[(size_t)(t * MG + b) * (3 * DD) + 2 * DD + h * 64 + tid];
    o[(size_t)(s * MG + b) * DD + h * 64 + tid] = accv * inv;
}

// ------------------------- residual + LayerNorm(512) -----------------------
__global__ __launch_bounds__(256)
void ln_res_kernel(float* __restrict__ x, const float* __restrict__ y,
                   const float* __restrict__ g, const float* __restrict__ b)
{
    const int row = blockIdx.x, tid = threadIdx.x;
    __shared__ float rs[256], rq[256];
    const size_t base = (size_t)row * DD;
    float v0 = x[base + tid] + y[base + tid];
    float v1 = x[base + 256 + tid] + y[base + 256 + tid];
    rs[tid] = v0 + v1; rq[tid] = v0 * v0 + v1 * v1;
    __syncthreads();
    for (int off = 128; off > 0; off >>= 1) {
        if (tid < off) { rs[tid] += rs[tid + off]; rq[tid] += rq[tid + off]; }
        __syncthreads();
    }
    float mu = rs[0] * (1.f / DD);
    float var = rq[0] * (1.f / DD) - mu * mu;
    float inv = rsqrtf(var + 1e-5f);
    x[base + tid]       = (v0 - mu) * inv * g[tid]       + b[tid];
    x[base + 256 + tid] = (v1 - mu) * inv * g[256 + tid] + b[256 + tid];
}

// ------------------------- GCN helpers -------------------------------------
__global__ void fill_f32_kernel(float* p, float v, int n)
{ int i = blockIdx.x * 256 + threadIdx.x; if (i < n) p[i] = v; }

__global__ void fill_i32_kernel(int* p, int v, int n)
{ int i = blockIdx.x * 256 + threadIdx.x; if (i < n) p[i] = v; }

__global__ void deg_edges_kernel(float* __restrict__ deg, const int* __restrict__ dst,
                                 const int* __restrict__ mask, int E)
{
    int e = blockIdx.x * 256 + threadIdx.x;
    if (e < E && (!mask || mask[e])) atomicAdd(&deg[dst[e]], 1.0f);
}

__global__ void rsqrt_inplace_kernel(float* p, int n)
{ int i = blockIdx.x * 256 + threadIdx.x; if (i < n) p[i] = rsqrtf(p[i]); }

__global__ __launch_bounds__(256)
void gcn_self_kernel(float* __restrict__ out, const float* __restrict__ hW,
                     const float* __restrict__ dis, const float* __restrict__ bias, int D)
{
    const int i = blockIdx.x;
    const float d2 = dis[i] * dis[i];
    for (int j = threadIdx.x; j < D; j += 256)
        out[(size_t)i * D + j] = hW[(size_t)i * D + j] * d2 + bias[j];
}

__global__ __launch_bounds__(256)
void gcn_scatter_kernel(float* __restrict__ out, const float* __restrict__ hW,
                        const float* __restrict__ dis, const int* __restrict__ src,
                        const int* __restrict__ dst, const int* __restrict__ mask, int D)
{
    const int e = blockIdx.x;
    if (mask && !mask[e]) return;
    const int s = src[e], d = dst[e];
    const float c = dis[s] * dis[d];
    for (int j = threadIdx.x; j < D; j += 256)
        atomicAdd(&out[(size_t)d * D + j], c * hW[(size_t)s * D + j]);
}

__global__ void relu_kernel(float* x, int n)
{ int i = blockIdx.x * 256 + threadIdx.x; if (i < n) x[i] = fmaxf(x[i], 0.f); }

// GEMV for the pool score projections (D=512 -> 1), one wave per row
__global__ __launch_bounds__(32)
void gemv_kernel(const float* __restrict__ h, const float* __restrict__ w,
                 float* __restrict__ out, int rows)
{
    const int r = blockIdx.x, lane = threadIdx.x;
    float s = 0.f;
    for (int j = lane; j < DD; j += 32) s += h[(size_t)r * DD + j] * w[j];
    for (int off = 16; off > 0; off >>= 1) s += __shfl_xor(s, off, 32);
    if (lane == 0) out[r] = s;
}

__global__ void score_self_kernel(float* __restrict__ sc, const float* __restrict__ sW,
                                  const float* __restrict__ dis, const float* __restrict__ pb, int n)
{
    int i = blockIdx.x * 256 + threadIdx.x;
    if (i < n) sc[i] = sW[i] * dis[i] * dis[i] + pb[0];
}

__global__ void score_scatter_kernel(float* __restrict__ sc, const float* __restrict__ sW,
                                     const float* __restrict__ dis, const int* __restrict__ src,
                                     const int* __restrict__ dst, const int* __restrict__ mask, int E)
{
    int e = blockIdx.x * 256 + threadIdx.x;
    if (e < E && (!mask || mask[e]))
        atomicAdd(&sc[dst[e]], dis[src[e]] * dis[dst[e]] * sW[src[e]]);
}

// per-graph exact top-k by rank counting (ties broken by smaller index, ==
// jax top_k order); rank < k  <=>  kept at slot g*k + rank
__global__ __launch_bounds__(64)
void topk_kernel(const float* __restrict__ score, int* __restrict__ nmap,
                 int* __restrict__ perm, float* __restrict__ gate, int n_per, int k)
{
    const int g = blockIdx.x, v = threadIdx.x;
    __shared__ float sc[64];
    sc[v] = (v < n_per) ? score[g * n_per + v] : -1e30f;
    __syncthreads();
    if (v < n_per) {
        const float sv = sc[v];
        int rank = 0;
        for (int u = 0; u < n_per; ++u) {
            float su = sc[u];
            if (su > sv || (su == sv && u < v)) ++rank;
        }
        if (rank < k) {
            int old = g * n_per + v;
            int ni  = g * k + rank;
            nmap[old] = ni;
            perm[ni]  = old;
            gate[ni]  = tanhf(sv);
        }
    }
}

__global__ __launch_bounds__(256)
void gather_scale_kernel(float* __restrict__ xn, const float* __restrict__ h,
                         const int* __restrict__ perm, const float* __restrict__ gate, int D)
{
    const int i = blockIdx.x;
    const int old = perm[i];
    const float gv = gate[i];
    for (int j = threadIdx.x; j < D; j += 256)
        xn[(size_t)i * D + j] = h[(size_t)old * D + j] * gv;
}

__global__ void edge_remap_kernel(int* __restrict__ s1, int* __restrict__ d1,
                                  int* __restrict__ m1, const int* __restrict__ src,
                                  const int* __restrict__ dst, const int* __restrict__ mask,
                                  const int* __restrict__ nmap, int E)
{
    int e = blockIdx.x * 256 + threadIdx.x;
    if (e >= E) return;
    int ok = (!mask || mask[e]);
    int ns = ok ? nmap[src[e]] : -1;
    int nd = ok ? nmap[dst[e]] : -1;
    if (ns >= 0 && nd >= 0) { s1[e] = ns; d1[e] = nd; m1[e] = 1; }
    else                    { s1[e] = 0;  d1[e] = 0;  m1[e] = 0; }
}

__global__ __launch_bounds__(256)
void readout_kernel(float* __restrict__ out, const float* __restrict__ h, int k, int D)
{
    const int g = blockIdx.x;
    for (int j = threadIdx.x; j < D; j += 256) {
        float mx = -1e30f, sm = 0.f;
        for (int i = 0; i < k; ++i) {
            float v = h[(size_t)(g * k + i) * D + j];
            mx = fmaxf(mx, v); sm += v;
        }
        out[(size_t)g * 2 * D + j]     = mx;
        out[(size_t)g * 2 * D + D + j] = sm / (float)k;
    }
}

__global__ void add_kernel(float* __restrict__ o, const float* __restrict__ a,
                           const float* __restrict__ b, int n)
{ int i = blockIdx.x * 256 + threadIdx.x; if (i < n) o[i] = a[i] + b[i]; }

__global__ __launch_bounds__(32)
void fc3_kernel(const float* __restrict__ z, const float* __restrict__ W,
                const float* __restrict__ b, float* __restrict__ out)
{
    const int g = blockIdx.x, lane = threadIdx.x;
    float s0 = 0.f, s1 = 0.f;
    for (int kk = lane; kk < 256; kk += 32) {
        float zv = z[g * 256 + kk];
        s0 += zv * W[kk * 2 + 0];
        s1 += zv * W[kk * 2 + 1];
    }
    for (int off = 16; off > 0; off >>= 1) {
        s0 += __shfl_xor(s0, off, 32);
        s1 += __shfl_xor(s1, off, 32);
    }
    if (lane == 0) { out[g * 2] = s0 + b[0]; out[g * 2 + 1] = s1 + b[1]; }
}

// ---------------------------------------------------------------------------
extern "C" void kernel_launch(void* const* d_in, const int* in_sizes, int n_in,
                              void* d_out, int out_size, void* d_ws, size_t ws_size,
                              hipStream_t stream)
{
    (void)in_sizes; (void)n_in; (void)out_size; (void)ws_size;

    const float* x      = (const float*)d_in[0];
    const int*   eidx   = (const int*)  d_in[1];
    const float* bn_g   = (const float*)d_in[2];
    const float* bn_b   = (const float*)d_in[3];
    const float* W_emb  = (const float*)d_in[4];
    const float* Wqkv   = (const float*)d_in[5];
    const float* bqkv   = (const float*)d_in[6];
    const float* Wo     = (const float*)d_in[7];
    const float* bo     = (const float*)d_in[8];
    const float* W1     = (const float*)d_in[9];
    const float* b1     = (const float*)d_in[10];
    const float* W2     = (const float*)d_in[11];
    const float* b2     = (const float*)d_in[12];
    const float* g1     = (const float*)d_in[13];
    const float* be1    = (const float*)d_in[14];
    const float* g2     = (const float*)d_in[15];
    const float* be2    = (const float*)d_in[16];
    const float* W_emb1 = (const float*)d_in[17];
    const float* c1W    = (const float*)d_in[18];
    const float* c1b    = (const float*)d_in[19];
    const float* csW    = (const float*)d_in[20];
    const float* csb    = (const float*)d_in[21];
    const float* p1W    = (const float*)d_in[22];
    const float* p1b    = (const float*)d_in[23];
    const float* p2W    = (const float*)d_in[24];
    const float* p2b    = (const float*)d_in[25];
    const float* f1W    = (const float*)d_in[26];
    const float* f1b    = (const float*)d_in[27];
    const float* f2W    = (const float*)d_in[28];
    const float* f2b    = (const float*)d_in[29];
    const float* f3W    = (const float*)d_in[30];
    const float* f3b    = (const float*)d_in[31];
    const int* src = eidx;
    const int* dst = eidx + EG;
    float* outp = (float*)d_out;

    // -------- workspace layout (float units); total ~38M floats (~152 MB) --
    float* ws = (float*)d_ws;
    size_t off = 0;
    float* T    = ws + off; off += (size_t)NTOK * DD;      // seq activations
    float* BIG  = ws + off; off += (size_t)NTOK * DFF_;    // qkv / ffn-hid / hW
    float* BA   = ws + off; off += (size_t)NTOK * DD;      // o / xg / conv ping
    float* BB   = ws + off; off += (size_t)NTOK * DD;      // tmp / h0 / pool2 out
    float* mu    = ws + off; off += NN_;
    float* rsd   = ws + off; off += NN_;
    float* dis0  = ws + off; off += NTOK;
    float* dis1  = ws + off; off += N1_;
    float* sWb   = ws + off; off += NTOK;
    float* scb   = ws + off; off += NTOK;
    int*   nmap  = (int*)(ws + off); off += NTOK;
    int*   perm1 = (int*)(ws + off); off += N1_;
    float* gate1 = ws + off; off += N1_;
    int*   nmap2 = (int*)(ws + off); off += N1_;
    int*   perm2 = (int*)(ws + off); off += N2_;
    float* gate2 = ws + off; off += N2_;
    int*   s1    = (int*)(ws + off); off += EG;
    int*   d1    = (int*)(ws + off); off += EG;
    int*   m1    = (int*)(ws + off); off += EG;
    float* x1    = ws + off; off += (size_t)MG * 2 * DD;
    float* x2    = ws + off; off += (size_t)MG * 2 * DD;
    float* zb    = ws + off; off += (size_t)MG * 2 * DD;
    float* t1    = ws + off; off += (size_t)MG * DD;
    float* t2    = ws + off; off += (size_t)MG * 256;
    bf16_t* Abf = (bf16_t*)(ws + off); off += (size_t)NTOK * DFF_ / 2;  // bf16 A
    bf16_t* Bbf = (bf16_t*)(ws + off); off += (size_t)DFF_ * DD / 2;    // bf16 B^T

    auto gemm = [&](const float* A, const float* Bm, const float* bias, float* C,
                    int M_, int N_, int K_, int relu) {
        if ((M_ % 64 == 0) && (N_ % 64 == 0) && (K_ % 32 == 0)) {
            const int n4 = M_ * K_ / 4;
            cvt_bf16_kernel<<<(n4 + 255) / 256, 256, 0, stream>>>(A, Abf, n4);
            cvt_bf16_T_kernel<<<(N_ * K_ + 255) / 256, 256, 0, stream>>>(Bm, Bbf, K_, N_);
            dim3 g(N_ / 64, M_ / 64);
            wmma_gemm64_kernel<<<g, 128, 0, stream>>>(Abf, Bbf, bias, C, M_, N_, K_, relu);
        } else {
            dim3 g((N_ + 31) / 32, (M_ + 31) / 32);
            wmma_gemm_edge_kernel<<<g, 128, 0, stream>>>(A, Bm, bias, C, M_, N_, K_, relu);
        }
    };
    const int EB = (EG + 255) / 256;

    // ---------------- phase A: BN + embedding + transformer -----------------
    bn_stats_kernel<<<NN_, 256, 0, stream>>>(x, mu, rsd);
    bn_apply_kernel<<<(NTOK * NN_ + 255) / 256, 256, 0, stream>>>(x, mu, rsd, bn_g, bn_b, BIG);
    gemm(BIG, W_emb, nullptr, T, NTOK, DD, NN_, 0);        // T = Axb @ W_emb  (edge: K=62)

    for (int l = 0; l < LL; ++l) {
        gemm(T, Wqkv + (size_t)l * DD * 3 * DD, bqkv + l * 3 * DD, BIG, NTOK, 3 * DD, DD, 0);
        attn_kernel<<<dim3(NN_, MG, HH), 64, 0, stream>>>(BIG, BA);
        gemm(BA, Wo + (size_t)l * DD * DD, bo + l * DD, BB, NTOK, DD, DD, 0);
        ln_res_kernel<<<NTOK, 256, 0, stream>>>(T, BB, g1 + l * DD, be1 + l * DD);
        gemm(T, W1 + (size_t)l * DD * DFF_, b1 + l * DFF_, BIG, NTOK, DFF_, DD, 1);
        gemm(BIG, W2 + (size_t)l * DFF_ * DD, b2 + l * DD, BB, NTOK, DD, DFF_, 0);
        ln_res_kernel<<<NTOK, 256, 0, stream>>>(T, BB, g2 + l * DD, be2 + l * DD);
    }

    // ---------------- phase B: GCN / SAGPool / readout ----------------------
    gemm(T, W_emb1, nullptr, BA, NTOK, NN_, DD, 0);        // xg in BA (edge: N=62)

    fill_f32_kernel<<<(NTOK + 255) / 256, 256, 0, stream>>>(dis0, 1.0f, NTOK);
    deg_edges_kernel<<<EB, 256, 0, stream>>>(dis0, dst, nullptr, EG);
    rsqrt_inplace_kernel<<<(NTOK + 255) / 256, 256, 0, stream>>>(dis0, NTOK);

    // conv1 + relu -> h0 in BB
    gemm(BA, c1W, nullptr, BIG, NTOK, DD, NN_, 0);         // edge: K=62
    gcn_self_kernel<<<NTOK, 256, 0, stream>>>(BB, BIG, dis0, c1b, DD);
    gcn_scatter_kernel<<<EG, 256, 0, stream>>>(BB, BIG, dis0, src, dst, nullptr, DD);
    relu_kernel<<<(NTOK * DD + 255) / 256, 256, 0, stream>>>(BB, NTOK * DD);

    // pool1 score + top-k
    gemv_kernel<<<NTOK, 32, 0, stream>>>(BB, p1W, sWb, NTOK);
    score_self_kernel<<<(NTOK + 255) / 256, 256, 0, stream>>>(scb, sWb, dis0, p1b, NTOK);
    score_scatter_kernel<<<EB, 256, 0, stream>>>(scb, sWb, dis0, src, dst, nullptr, EG);
    fill_i32_kernel<<<(NTOK + 255) / 256, 256, 0, stream>>>(nmap, -1, NTOK);
    topk_kernel<<<MG, 64, 0, stream>>>(scb, nmap, perm1, gate1, NN_, KP1);
    gather_scale_kernel<<<N1_, 256, 0, stream>>>(T, BB, perm1, gate1, DD);   // h1 in T
    edge_remap_kernel<<<EB, 256, 0, stream>>>(s1, d1, m1, src, dst, nullptr, nmap, EG);
    readout_kernel<<<MG, 256, 0, stream>>>(x1, T, KP1, DD);

    fill_f32_kernel<<<(N1_ + 255) / 256, 256, 0, stream>>>(dis1, 1.0f, N1_);
    deg_edges_kernel<<<EB, 256, 0, stream>>>(dis1, d1, m1, EG);
    rsqrt_inplace_kernel<<<(N1_ + 255) / 256, 256, 0, stream>>>(dis1, N1_);

    // 3 masked GCN layers, ping-pong T <-> BA
    float* hcur = T; float* hnext = BA;
    for (int l = 0; l < LL - 1; ++l) {
        gemm(hcur, csW + (size_t)l * DD * DD, nullptr, BIG, N1_, DD, DD, 0);
        gcn_self_kernel<<<N1_, 256, 0, stream>>>(hnext, BIG, dis1, csb + l * DD, DD);
        gcn_scatter_kernel<<<EG, 256, 0, stream>>>(hnext, BIG, dis1, s1, d1, m1, DD);
        relu_kernel<<<(N1_ * DD + 255) / 256, 256, 0, stream>>>(hnext, N1_ * DD);
        float* t_ = hcur; hcur = hnext; hnext = t_;
    }

    // pool2 (filtered adjacency unused downstream -> skipped)
    gemv_kernel<<<N1_, 32, 0, stream>>>(hcur, p2W, sWb, N1_);
    score_self_kernel<<<(N1_ + 255) / 256, 256, 0, stream>>>(scb, sWb, dis1, p2b, N1_);
    score_scatter_kernel<<<EB, 256, 0, stream>>>(scb, sWb, dis1, s1, d1, m1, EG);
    fill_i32_kernel<<<(N1_ + 255) / 256, 256, 0, stream>>>(nmap2, -1, N1_);
    topk_kernel<<<MG, 64, 0, stream>>>(scb, nmap2, perm2, gate2, KP1, KP2);
    gather_scale_kernel<<<N2_, 256, 0, stream>>>(BB, hcur, perm2, gate2, DD);
    readout_kernel<<<MG, 256, 0, stream>>>(x2, BB, KP2, DD);

    // ---------------- head --------------------------------------------------
    add_kernel<<<(MG * 2 * DD + 255) / 256, 256, 0, stream>>>(zb, x1, x2, MG * 2 * DD);
    gemm(zb, f1W, f1b, t1, MG, DD, 2 * DD, 1);
    gemm(t1, f2W, f2b, t2, MG, 256, DD, 1);
    fc3_kernel<<<MG, 32, 0, stream>>>(t2, f3W, f3b, outp);
}
